// FERNN_74998718923034
// MI455X (gfx1250) — matrix operand order repaired and checked by reading, not compiled, not run on hardware
//
#include <hip/hip_runtime.h>
#include <math.h>

// ---------------------------------------------------------------------------
// CDNA5 (gfx1250) wave32 WMMA implementation of the FERNN reference.
// Heavy 3x3 convs are implicit GEMMs on v_wmma_f32_16x16x32_bf16.
// Round 2: 2 M-tiles per wave (32 px x 64 oc) so each B-fragment group feeds
// 8 WMMAs -> half the weight-fetch traffic and better load-latency hiding.
// ---------------------------------------------------------------------------

typedef __attribute__((ext_vector_type(16))) __bf16 v16bf;
typedef __attribute__((ext_vector_type(8)))  __bf16 v8bf;
typedef __attribute__((ext_vector_type(8)))  float  v8f;

#define BN_SCALEF 0.99999500003749975f   // 1/sqrt(1 + 1e-5)

__device__ __forceinline__ v16bf cat8(v8bf lo, v8bf hi) {
  v16bf r;
#pragma unroll
  for (int i = 0; i < 8; ++i) { r[i] = lo[i]; r[i + 8] = hi[i]; }
  return r;
}

// ---------------------------------------------------------------------------
// Weight re-layout: w[o][c][ky][kx] (f32, OIHW) -> Wg[o][(ky*3+kx)*CIN + c] bf16
// A K-slice of 32 = one tap x 32 contiguous channels, so B-matrix fragments
// are two contiguous 16B loads per lane.
// ---------------------------------------------------------------------------
__global__ void wt_to_bf16(const float* __restrict__ w, __bf16* __restrict__ out,
                           int cin) {
  int idx = blockIdx.x * 256 + threadIdx.x;
  int total = 64 * cin * 9;
  if (idx >= total) return;
  int o   = idx / (cin * 9);
  int rem = idx % (cin * 9);
  int tap = rem / cin;
  int c   = rem % cin;
  out[(size_t)o * (9 * cin) + tap * cin + c] =
      (__bf16)w[((size_t)o * cin + c) * 9 + tap];
}

// ---------------------------------------------------------------------------
// Tiny first conv: (B=4) 3->32, 5x5, zero pad 2, BN + ReLU.  <0.2% of FLOPs.
// ---------------------------------------------------------------------------
__global__ __launch_bounds__(256) void conv1_5x5(
    const float* __restrict__ u, int t, const float* __restrict__ w,
    const float* __restrict__ g, const float* __restrict__ b,
    float* __restrict__ out) {
  int idx = blockIdx.x * 256 + threadIdx.x;  // 4*32*64*64 threads
  int x = idx & 63, y = (idx >> 6) & 63, o = (idx >> 12) & 31, bb = idx >> 17;
  const float* up = u + (size_t)(bb * 8 + t) * 3 * 4096;
  float acc = 0.f;
#pragma unroll
  for (int c = 0; c < 3; ++c)
    for (int dy = 0; dy < 5; ++dy) {
      int iy = y + dy - 2;
      if (iy < 0 || iy > 63) continue;
      for (int dx = 0; dx < 5; ++dx) {
        int ix = x + dx - 2;
        if (ix < 0 || ix > 63) continue;
        acc = fmaf(up[(c * 64 + iy) * 64 + ix],
                   w[((o * 3 + c) * 5 + dy) * 5 + dx], acc);
      }
    }
  out[idx] = fmaxf(fmaf(acc, g[o] * BN_SCALEF, b[o]), 0.f);
}

// ---------------------------------------------------------------------------
// Generic 3x3 conv as implicit GEMM on bf16 WMMA.
//   block = 256 threads = 8 wave32; grid.x = nimg * 16 (four output rows/block)
//   wave w: rows rp = (w>>2)*2 .. rp+1, x-segment x0 = (w&3)*16
//           -> 32 px x 64 oc per wave, 8 f32 accumulators.
//   CIN in {32,64}, COUT = 64, H = W = 64.
// Template:
//   CIRC_SHIFT : circular padding + per-image velocity shift (recurrent convs)
//   EPI        : 0 = BN(scale,bias)+ReLU     1 = tanh(acc + add)
//   ADD_BCAST  : add tensor indexed by img/9 (x broadcast over V)
//   IN_TSLICE  : input is a (v,b,c,t,y,x) slice of d_out at time tprev
//   OUT_TSLICE : output goes to the (v,b,c,t,y,x) slice of d_out at time t
// ---------------------------------------------------------------------------
template <int CIN, bool CIRC_SHIFT, int EPI, bool ADD_BCAST, bool IN_TSLICE,
          bool OUT_TSLICE>
__global__ __launch_bounds__(256) void conv3x3_wmma(
    const float* __restrict__ in, int zero_in, const __bf16* __restrict__ Wg,
    const float* __restrict__ gamma, const float* __restrict__ beta,
    const float* __restrict__ addt, float* __restrict__ out, int t, int tprev) {
  constexpr int LS = CIN + 8;   // LDS channel stride (bank spread, 16B aligned)
  constexpr int KT = 9 * CIN;   // GEMM K
  // 6 input rows (4 output rows + 2 halo) x 66(+2) x CIN, bf16
  __shared__ __align__(16) __bf16 tile[6 * 68 * LS];

  const int img  = blockIdx.x >> 4;
  const int y0   = blockIdx.x & 15;   // 4-row group
  const int tid  = threadIdx.x;
  const int lane = tid & 31;
  const int wave = tid >> 5;
  const int rp   = (wave >> 2) << 1;  // first of this wave's two output rows
  const int x0w  = (wave & 3) << 4;

  const v8f vz = {};
  v8f acc[2][4];
#pragma unroll
  for (int i = 0; i < 2; ++i)
#pragma unroll
    for (int j = 0; j < 4; ++j) acc[i][j] = vz;

  if (!zero_in) {  // uniform across grid -> EXEC stays all-ones for WMMA
    // -------- stage input halo tile to LDS as bf16 (coalesced in x) --------
    int vy = 0, vx = 0;
    if (CIRC_SHIFT) { int v = img % 9; vy = v / 3 - 1; vx = v % 3 - 1; }
#pragma unroll 1
    for (int idx = tid; idx < 6 * 66 * CIN; idx += 256) {
      int xi = idx % 66;
      int q  = idx / 66;
      int c  = q % CIN;
      int tr = q / CIN;
      int iny = 4 * y0 - 1 + tr + vy;
      int inx = xi - 1 + vx;
      float val = 0.f;
      if (CIRC_SHIFT) {
        iny = (iny + 64) & 63;
        inx = (inx + 64) & 63;
        size_t ii;
        if (IN_TSLICE) {
          int bb = img / 9, vv = img % 9;
          ii = ((((size_t)(vv * 4 + bb) * CIN + c) * 8 + tprev) * 64 + iny) * 64 + inx;
        } else {
          ii = (((size_t)img * CIN + c) * 64 + iny) * 64 + inx;
        }
        val = in[ii];
      } else {
        if (iny >= 0 && iny < 64 && inx >= 0 && inx < 64)
          val = in[(((size_t)img * CIN + c) * 64 + iny) * 64 + inx];
      }
      tile[(tr * 68 + xi) * LS + c] = (__bf16)val;
    }
    __syncthreads();

    // -------- implicit GEMM: 9 taps x (CIN/32) K-slices x 4 N-tiles -------
    const int m  = lane & 15;
    const int h8 = (lane >> 4) << 4;    // A: hi lanes take K offsets 16..31
    const int bk = (lane >> 4) << 4;    // B: hi lanes take K offsets 16..31
    const int oc = lane & 15;           // B/C column within 16-wide N tile
#pragma unroll
    for (int tap = 0; tap < 9; ++tap) {
      const int ky = tap / 3, kx = tap % 3;
#pragma unroll
      for (int cs = 0; cs < CIN; cs += 32) {
        const int ab0 = ((rp + ky) * 68 + (x0w + m + kx)) * LS + cs + h8;
        const int ab1 = ab0 + 68 * LS;  // next output row
        v16bf A0 = cat8(*(const v8bf*)(tile + ab0),
                        *(const v8bf*)(tile + ab0 + 8));
        v16bf A1 = cat8(*(const v8bf*)(tile + ab1),
                        *(const v8bf*)(tile + ab1 + 8));
        const __bf16* wp = Wg + (size_t)oc * KT + (tap * CIN + cs + bk);
        v16bf B0 = cat8(*(const v8bf*)(wp),           *(const v8bf*)(wp + 8));
        v16bf B1 = cat8(*(const v8bf*)(wp + 16 * KT), *(const v8bf*)(wp + 16 * KT + 8));
        v16bf B2 = cat8(*(const v8bf*)(wp + 32 * KT), *(const v8bf*)(wp + 32 * KT + 8));
        v16bf B3 = cat8(*(const v8bf*)(wp + 48 * KT), *(const v8bf*)(wp + 48 * KT + 8));
        acc[0][0] = __builtin_amdgcn_wmma_f32_16x16x32_bf16(false, A0, false, B0,
                                                            (short)0, acc[0][0], false, false);
        acc[1][0] = __builtin_amdgcn_wmma_f32_16x16x32_bf16(false, A1, false, B0,
                                                            (short)0, acc[1][0], false, false);
        acc[0][1] = __builtin_amdgcn_wmma_f32_16x16x32_bf16(false, A0, false, B1,
                                                            (short)0, acc[0][1], false, false);
        acc[1][1] = __builtin_amdgcn_wmma_f32_16x16x32_bf16(false, A1, false, B1,
                                                            (short)0, acc[1][1], false, false);
        acc[0][2] = __builtin_amdgcn_wmma_f32_16x16x32_bf16(false, A0, false, B2,
                                                            (short)0, acc[0][2], false, false);
        acc[1][2] = __builtin_amdgcn_wmma_f32_16x16x32_bf16(false, A1, false, B2,
                                                            (short)0, acc[1][2], false, false);
        acc[0][3] = __builtin_amdgcn_wmma_f32_16x16x32_bf16(false, A0, false, B3,
                                                            (short)0, acc[0][3], false, false);
        acc[1][3] = __builtin_amdgcn_wmma_f32_16x16x32_bf16(false, A1, false, B3,
                                                            (short)0, acc[1][3], false, false);
      }
    }
  }

  // -------- fused epilogue + store.  C layout: lane holds fixed column
  // o = obase + (lane&15), rows M = g + 8*(lane>=16) -> x contiguous in g.
  const int n  = lane & 15;
  const int xb = x0w + ((lane >> 4) << 3);

  auto epi_store = [&](v8f a, int obase, int y) {
    const int o = obase + n;
    float v[8];
#pragma unroll
    for (int g = 0; g < 8; ++g) v[g] = a[g];
    if constexpr (EPI == 0) {
      const float sc = gamma[o] * BN_SCALEF;
      const float bi = beta[o];
#pragma unroll
      for (int g = 0; g < 8; ++g) v[g] = fmaxf(fmaf(v[g], sc, bi), 0.f);
    } else {
      const size_t aimg = ADD_BCAST ? (size_t)(img / 9) : (size_t)img;
      const float* ap = addt + (((aimg * 64 + o) * 64 + y) * 64 + xb);
      const float4 a0 = *(const float4*)ap;
      const float4 a1 = *(const float4*)(ap + 4);
      v[0] = tanhf(v[0] + a0.x); v[1] = tanhf(v[1] + a0.y);
      v[2] = tanhf(v[2] + a0.z); v[3] = tanhf(v[3] + a0.w);
      v[4] = tanhf(v[4] + a1.x); v[5] = tanhf(v[5] + a1.y);
      v[6] = tanhf(v[6] + a1.z); v[7] = tanhf(v[7] + a1.w);
    }
    size_t oidx;
    if constexpr (OUT_TSLICE) {
      const int bb = img / 9, vv = img % 9;
      oidx = ((((size_t)(vv * 4 + bb) * 64 + o) * 8 + t) * 64 + y) * 64 + xb;
    } else {
      oidx = (((size_t)img * 64 + o) * 64 + y) * 64 + xb;
    }
    *(float4*)(out + oidx)     = make_float4(v[0], v[1], v[2], v[3]);
    *(float4*)(out + oidx + 4) = make_float4(v[4], v[5], v[6], v[7]);
  };
#pragma unroll
  for (int rr = 0; rr < 2; ++rr) {
    const int y = 4 * y0 + rp + rr;
    epi_store(acc[rr][0], 0,  y);
    epi_store(acc[rr][1], 16, y);
    epi_store(acc[rr][2], 32, y);
    epi_store(acc[rr][3], 48, y);
  }
}

// ---------------------------------------------------------------------------
// Host driver.  48 stream-ordered kernel launches (6 per timestep).
// h1 lives directly inside d_out (read slice t-1, write slice t).
// Workspace ~119 MB: bf16 weights, x1/x2/x3, up, h0 ping-pong.
// ---------------------------------------------------------------------------
extern "C" void kernel_launch(void* const* d_in, const int* in_sizes, int n_in,
                              void* d_out, int out_size, void* d_ws,
                              size_t ws_size, hipStream_t stream) {
  const float* u    = (const float*)d_in[0];
  const float* w_u1 = (const float*)d_in[1];
  const float* g1   = (const float*)d_in[2];
  const float* b1   = (const float*)d_in[3];
  const float* w_u2 = (const float*)d_in[4];
  const float* g2   = (const float*)d_in[5];
  const float* b2   = (const float*)d_in[6];
  const float* w_u3 = (const float*)d_in[7];
  const float* g3   = (const float*)d_in[8];
  const float* b3   = (const float*)d_in[9];
  const float* w_ul = (const float*)d_in[10];
  const float* gul  = (const float*)d_in[11];
  const float* bul  = (const float*)d_in[12];
  const float* w_h0 = (const float*)d_in[13];
  const float* w_h1 = (const float*)d_in[14];
  float* outp = (float*)d_out;

  char* ws = (char*)d_ws;
  size_t off = 0;
  auto take = [&](size_t bytes) -> void* {
    void* p = ws + off;
    off += (bytes + 255) & ~(size_t)255;
    return p;
  };
  __bf16* Wb2  = (__bf16*)take((size_t)64 * 288 * 2);
  __bf16* Wb3  = (__bf16*)take((size_t)64 * 576 * 2);
  __bf16* Wbul = (__bf16*)take((size_t)64 * 576 * 2);
  __bf16* Wbh0 = (__bf16*)take((size_t)64 * 576 * 2);
  __bf16* Wbh1 = (__bf16*)take((size_t)64 * 576 * 2);
  float* x1  = (float*)take((size_t)4 * 32 * 4096 * 4);
  float* x2  = (float*)take((size_t)4 * 64 * 4096 * 4);
  float* x3  = (float*)take((size_t)4 * 64 * 4096 * 4);
  float* upb = (float*)take((size_t)36 * 64 * 4096 * 4);
  float* h0a = (float*)take((size_t)36 * 64 * 4096 * 4);
  float* h0b = (float*)take((size_t)36 * 64 * 4096 * 4);

  wt_to_bf16<<<(64 * 288 + 255) / 256, 256, 0, stream>>>(w_u2, Wb2, 32);
  wt_to_bf16<<<(64 * 576 + 255) / 256, 256, 0, stream>>>(w_u3, Wb3, 64);
  wt_to_bf16<<<(64 * 576 + 255) / 256, 256, 0, stream>>>(w_ul, Wbul, 64);
  wt_to_bf16<<<(64 * 576 + 255) / 256, 256, 0, stream>>>(w_h0, Wbh0, 64);
  wt_to_bf16<<<(64 * 576 + 255) / 256, 256, 0, stream>>>(w_h1, Wbh1, 64);

  float* h0p = h0a;
  float* h0n = h0b;
  for (int t = 0; t < 8; ++t) {
    conv1_5x5<<<2048, 256, 0, stream>>>(u, t, w_u1, g1, b1, x1);
    // 32->64 zero-pad, BN+ReLU
    conv3x3_wmma<32, false, 0, false, false, false><<<4 * 16, 256, 0, stream>>>(
        x1, 0, Wb2, g2, b2, nullptr, x2, 0, 0);
    // 64->64 zero-pad, BN+ReLU  -> x
    conv3x3_wmma<64, false, 0, false, false, false><<<4 * 16, 256, 0, stream>>>(
        x2, 0, Wb3, g3, b3, nullptr, x3, 0, 0);
    // h0 = tanh(x[:,None] + circ-conv(shift(h0)))
    conv3x3_wmma<64, true, 1, true, false, false><<<36 * 16, 256, 0, stream>>>(
        h0p, t == 0 ? 1 : 0, Wbh0, nullptr, nullptr, x3, h0n, 0, 0);
    // up = relu(bn(conv(h0)))
    conv3x3_wmma<64, false, 0, false, false, false><<<36 * 16, 256, 0, stream>>>(
        h0n, 0, Wbul, gul, bul, nullptr, upb, 0, 0);
    // h1 = tanh(up + circ-conv(shift(h1)));  h1 lives in d_out slices
    conv3x3_wmma<64, true, 1, false, true, true><<<36 * 16, 256, 0, stream>>>(
        outp, t == 0 ? 1 : 0, Wbh1, nullptr, nullptr, upb, outp, t, t - 1);
    float* tmp = h0p; h0p = h0n; h0n = tmp;
  }
  (void)in_sizes; (void)n_in; (void)out_size; (void)ws_size;
}
